// RNN_12670153523827
// MI455X (gfx1250) — compile-verified, matrix operand order checked by reading, not run on hardware
//
#include <hip/hip_runtime.h>
#include <hip/hip_bf16.h>
#include <stdint.h>

// ---------------------------------------------------------------------------
// 2-layer LSTM (B=256, T=1024, D=64, H=128) + FC on last step, for gfx1250.
//  * batch-parallel: 8 WGs x 32 batch rows; whole time loop in-kernel
//  * fused gate GEMM per step: [32 x K] @ [K x 512] via v_wmma_f32_16x16x32_f16
//  * weights as f16 WMMA B-fragments RESIDENT IN VGPRS for all 1024 steps
//  * c-state in VGPRs; h-state recycled in a small LDS activation tile
//  * layer-1 staging via global_load_async_to_lds_b128 (ASYNCcnt)
// ---------------------------------------------------------------------------

typedef __attribute__((ext_vector_type(16))) _Float16 v16h;
typedef __attribute__((ext_vector_type(8)))  float    v8f;

#define B_   256
#define T_   1024
#define D_   64
#define H_   128
#define G_   512      // 4*H
#define ROWS 32       // batch rows per workgroup
#define K0P  (D_ + H_ + 8)   // 200 halves padded pitch (layer 0 pack)
#define K1P  (H_ + H_ + 8)   // 264 halves padded pitch (layer 1 pack)

union Frag32 { uint4 q[2]; v16h v; };
union H8     { uint4 q;    _Float16 h[8]; };

static __device__ __forceinline__ float sigmoidf_(float x) {
  return 1.0f / (1.0f + __expf(-x));
}
static __device__ __forceinline__ float tanhf_(float x) {
  return 2.0f / (1.0f + __expf(-2.0f * x)) - 1.0f;   // saturates cleanly
}

// 16x32 f16 fragment load (documented CDNA5 layout): lane<16 holds
// K = {ko..ko+7, ko+16..ko+23}; lanes 16..31 shift base by +8 halves.
static __device__ __forceinline__ v16h ld_frag(const _Float16* p, int ko) {
  Frag32 u;
  u.q[0] = *(const uint4*)(p + ko);
  u.q[1] = *(const uint4*)(p + ko + 16);
  return u.v;
}

// async global -> LDS copy, 16B per lane, tracked by ASYNCcnt
static __device__ __forceinline__ void async_g2l_b128(uint32_t lds_off,
                                                      const void* gptr) {
  asm volatile("global_load_async_to_lds_b128 %0, %1, off"
               :: "v"(lds_off), "v"((uint64_t)(uintptr_t)gptr)
               : "memory");
}

// ---------------------------------------------------------------------------
// Pack f32 weights -> f16, concatenating [w_ih | w_hh] per gate row, padded.
// ---------------------------------------------------------------------------
__global__ __launch_bounds__(64) void pack_kernel(
    const float* __restrict__ w_ih0, const float* __restrict__ w_hh0,
    const float* __restrict__ b0,
    const float* __restrict__ w_ih1, const float* __restrict__ w_hh1,
    const float* __restrict__ b1,
    _Float16* __restrict__ w0p, _Float16* __restrict__ w1p,
    float* __restrict__ b0f, float* __restrict__ b1f)
{
  const int g = blockIdx.x;           // 0..511 gate row
  const int t = threadIdx.x;          // 0..63
  for (int k = t; k < K0P; k += 64) {
    _Float16 v = (_Float16)0.0f;
    if (k < D_)            v = (_Float16)w_ih0[g * D_ + k];
    else if (k < D_ + H_)  v = (_Float16)w_hh0[g * H_ + (k - D_)];
    w0p[g * K0P + k] = v;
  }
  for (int k = t; k < K1P; k += 64) {
    _Float16 v = (_Float16)0.0f;
    if (k < H_)            v = (_Float16)w_ih1[g * H_ + k];
    else if (k < 2 * H_)   v = (_Float16)w_hh1[g * H_ + (k - H_)];
    w1p[g * K1P + k] = v;
  }
  if (t == 0) { b0f[g] = b0[g]; b1f[g] = b1[g]; }
}

// ---------------------------------------------------------------------------
// One LSTM layer, full time loop. grid = B/ROWS workgroups of 256 threads.
//   L0=true : xin = f32 [B,T,D], writes hseq_out f16 [B,T,H]
//   L0=false: hin = f16 [B,T,H], writes hlast_out f32 [B,H]
// Wave wv owns gate N-tiles {wv, wv+8, wv+16, wv+24} == hidden cols
// 16wv..16wv+15 of gates i,f,g,o -> gate math needs no cross-wave exchange.
// ---------------------------------------------------------------------------
template<int KX, bool L0>
__global__ __launch_bounds__(256) void lstm_layer_kernel(
    const float*    __restrict__ xin,
    const _Float16* __restrict__ hin,
    const _Float16* __restrict__ wpack,   // [G, KP] f16 (padded rows)
    const float*    __restrict__ bias,    // [G] f32
    const float*    __restrict__ h0,      // [B, H] f32 (this layer's slice)
    _Float16*       __restrict__ hseq_out,
    float*          __restrict__ hlast_out)
{
  constexpr int KT     = KX + H_;          // 192 or 256 (mult of 32)
  constexpr int KP     = KT + 8;           // padded pitch in halves
  constexpr int KSTEPS = KT / 32;          // 6 or 8

  extern __shared__ _Float16 sA[];         // [ROWS][KP]  cols: [x_t | h]

  const int tid    = threadIdx.x;
  const int lane   = tid & 31;
  const int wv     = tid >> 5;             // 0..7
  const int l16    = lane & 15;
  const int hi16   = lane >> 4;            // 0/1
  const int kbase  = hi16 * 8;             // A/B fragment K base
  const int rowoff = hi16 * 8;             // C/D fragment M offset
  const int bbase  = blockIdx.x * ROWS;

  // ---- weight B-fragments: load once from global, resident in VGPRs ----
  v16h bfr[KSTEPS][4];
#pragma unroll
  for (int kk = 0; kk < KSTEPS; ++kk) {
#pragma unroll
    for (int j = 0; j < 4; ++j) {
      const int grow = j * H_ + wv * 16 + l16;     // W row == output column
      const _Float16* p = wpack + (size_t)grow * KP + kk * 32 + kbase;
      Frag32 u;
      u.q[0] = *(const uint4*)p;
      u.q[1] = *(const uint4*)(p + 16);
      bfr[kk][j] = u.v;
    }
  }

  // ---- init h columns of activation tile from h0 ----
  for (int i = tid; i < ROWS * H_; i += 256) {
    int r = i >> 7, c = i & (H_ - 1);
    sA[r * KP + KX + c] = (_Float16)h0[(bbase + r) * H_ + c];
  }

  // per-lane gate biases: gate j lives at columns j*128 + 16*wv + l16
  float bj[4];
#pragma unroll
  for (int j = 0; j < 4; ++j) bj[j] = bias[j * H_ + wv * 16 + l16];

  float cst[2][8];
#pragma unroll
  for (int m = 0; m < 2; ++m)
#pragma unroll
    for (int r = 0; r < 8; ++r) cst[m][r] = 0.0f;

  const _Float16* arow0 = sA + l16 * KP;          // A fragment, M-tile 0
  const _Float16* arow1 = sA + (16 + l16) * KP;   // A fragment, M-tile 1

  for (int t = 0; t < T_; ++t) {
    // ---- stage x_t into cols [0, KX) of sA ----
    const int r = tid >> 3, ch = tid & 7;
    if constexpr (L0) {                           // f32 -> f16 convert
      const float* sp = xin + (((size_t)(bbase + r)) * T_ + t) * D_ + ch * 8;
      float4 a  = *(const float4*)sp;
      float4 b2 = *(const float4*)(sp + 4);
      H8 u;
      u.h[0] = (_Float16)a.x;  u.h[1] = (_Float16)a.y;
      u.h[2] = (_Float16)a.z;  u.h[3] = (_Float16)a.w;
      u.h[4] = (_Float16)b2.x; u.h[5] = (_Float16)b2.y;
      u.h[6] = (_Float16)b2.z; u.h[7] = (_Float16)b2.w;
      *(uint4*)(sA + r * KP + ch * 8) = u.q;
      if (t + 1 < T_) __builtin_prefetch(sp + D_, 0, 0);
    } else {                                      // pure f16 copy: async DMA
      const _Float16* sp = hin + (((size_t)(bbase + r)) * T_ + t) * H_ + ch * 16;
      const uint32_t loff =
          (uint32_t)(uintptr_t)(sA + r * KP + ch * 16);   // low 32 = LDS addr
      async_g2l_b128(loff, sp);
      async_g2l_b128(loff + 16, sp + 8);
      asm volatile("s_wait_asynccnt 0x0" ::: "memory");
      if (t + 1 < T_) __builtin_prefetch(sp + H_, 0, 0);
    }
    __syncthreads();   // staging + previous step's h-writes visible

    // ---- gate GEMM: g = [x_t | h] @ W^T (B resident in VGPRs) ----
    v8f acc[2][4];
#pragma unroll
    for (int m = 0; m < 2; ++m)
#pragma unroll
      for (int j = 0; j < 4; ++j) acc[m][j] = (v8f){};

#pragma unroll
    for (int kk = 0; kk < KSTEPS; ++kk) {
      const int ko = kk * 32 + kbase;
      v16h a0 = ld_frag(arow0, ko);
      v16h a1 = ld_frag(arow1, ko);
#pragma unroll
      for (int j = 0; j < 4; ++j) {
        acc[0][j] = __builtin_amdgcn_wmma_f32_16x16x32_f16(
            false, a0, false, bfr[kk][j], (short)0, acc[0][j], false, false);
        acc[1][j] = __builtin_amdgcn_wmma_f32_16x16x32_f16(
            false, a1, false, bfr[kk][j], (short)0, acc[1][j], false, false);
      }
    }
    __syncthreads();   // all LDS reads of sA done before h rewrite

    // ---- gate nonlinearity + state update ----
    const int col = wv * 16 + l16;
#pragma unroll
    for (int m = 0; m < 2; ++m) {
#pragma unroll
      for (int rr = 0; rr < 8; ++rr) {
        float gi = acc[m][0][rr] + bj[0];
        float gf = acc[m][1][rr] + bj[1];
        float gg = acc[m][2][rr] + bj[2];
        float go = acc[m][3][rr] + bj[3];
        float c  = sigmoidf_(gf) * cst[m][rr] + sigmoidf_(gi) * tanhf_(gg);
        float h  = sigmoidf_(go) * tanhf_(c);
        cst[m][rr] = c;
        const int row = m * 16 + rowoff + rr;      // local batch row 0..31
        const _Float16 hh = (_Float16)h;
        sA[row * KP + KX + col] = hh;              // next step's A operand
        if constexpr (L0)
          hseq_out[(((size_t)(bbase + row)) * T_ + t) * H_ + col] = hh;
        if constexpr (!L0)
          if (t == T_ - 1) hlast_out[(bbase + row) * H_ + col] = h;
      }
    }
    // next staging writes x-cols (disjoint from h-cols); top barrier orders.
  }
}

// ---------------------------------------------------------------------------
// Final FC: out[b] = h1_last[b,:] . fc_w + fc_b
// ---------------------------------------------------------------------------
__global__ __launch_bounds__(256) void fc_kernel(
    const float* __restrict__ hlast, const float* __restrict__ fcw,
    const float* __restrict__ fcb, float* __restrict__ out)
{
  const int b = blockIdx.x * blockDim.x + threadIdx.x;
  if (b >= B_) return;
  const float4* hp = (const float4*)(hlast + b * H_);
  const float4* wp = (const float4*)fcw;
  float s = 0.0f;
#pragma unroll
  for (int i = 0; i < H_ / 4; ++i) {
    float4 h4 = hp[i], w4 = wp[i];
    s += h4.x * w4.x + h4.y * w4.y + h4.z * w4.z + h4.w * w4.w;
  }
  out[b] = s + fcb[0];
}

// ---------------------------------------------------------------------------
extern "C" void kernel_launch(void* const* d_in, const int* in_sizes, int n_in,
                              void* d_out, int out_size, void* d_ws, size_t ws_size,
                              hipStream_t stream) {
  const float* x     = (const float*)d_in[0];   // [B,T,D]
  const float* h0    = (const float*)d_in[1];   // [L,B,H]
  const float* w_ih0 = (const float*)d_in[2];
  const float* w_hh0 = (const float*)d_in[3];
  const float* b0    = (const float*)d_in[4];
  const float* w_ih1 = (const float*)d_in[5];
  const float* w_hh1 = (const float*)d_in[6];
  const float* b1    = (const float*)d_in[7];
  const float* fcw   = (const float*)d_in[8];
  const float* fcb   = (const float*)d_in[9];

  char* ws = (char*)d_ws;
  size_t off = 0;
  auto carve = [&](size_t bytes) -> void* {
    void* p = ws + off;
    off = (off + bytes + 255) & ~(size_t)255;
    return p;
  };
  _Float16* w0p    = (_Float16*)carve((size_t)G_ * K0P * sizeof(_Float16));
  _Float16* w1p    = (_Float16*)carve((size_t)G_ * K1P * sizeof(_Float16));
  float*    b0f    = (float*)carve(G_ * sizeof(float));
  float*    b1f    = (float*)carve(G_ * sizeof(float));
  _Float16* hseq   = (_Float16*)carve((size_t)B_ * T_ * H_ * sizeof(_Float16));
  float*    h1last = (float*)carve((size_t)B_ * H_ * sizeof(float));

  pack_kernel<<<G_, 64, 0, stream>>>(w_ih0, w_hh0, b0, w_ih1, w_hh1, b1,
                                     w0p, w1p, b0f, b1f);

  const size_t lds0 = (size_t)ROWS * (D_ + H_ + 8) * sizeof(_Float16); // ~12.5 KB
  lstm_layer_kernel<D_, true><<<B_ / ROWS, 256, lds0, stream>>>(
      x, nullptr, w0p, b0f, h0, hseq, nullptr);

  const size_t lds1 = (size_t)ROWS * (2 * H_ + 8) * sizeof(_Float16);  // ~16.5 KB
  lstm_layer_kernel<H_, false><<<B_ / ROWS, 256, lds1, stream>>>(
      nullptr, hseq, w1p, b1f, h0 + (size_t)B_ * H_, nullptr, h1last);

  fc_kernel<<<1, 256, 0, stream>>>(h1last, fcw, fcb, (float*)d_out);
}
